// MistralAttention_8907762172133
// MI455X (gfx1250) — matrix operand order repaired and checked by reading, not compiled, not run on hardware
//
#include <hip/hip_runtime.h>
#include <cstdint>
#include <cstddef>

typedef __bf16 bf16;
typedef __attribute__((ext_vector_type(16))) __bf16 v16bf;
typedef __attribute__((ext_vector_type(8)))  __bf16 v8bf;
typedef __attribute__((ext_vector_type(8)))  float  v8f;

#define B_   2
#define S_   2048
#define H_   4096
#define NH   32
#define NKV  8
#define HD   128
#define WIN  1024

// ---------------------------------------------------------------------------
// A-matrix fragment (16-bit, 16x32): lane l holds row m=l%16.
// a[0..15] = K (16g .. 16g+15),  g = l/16.
// ---------------------------------------------------------------------------
__device__ inline v16bf load_a_frag(const bf16* p) {
    v8bf lo = *(const v8bf*)(p);
    v8bf hi = *(const v8bf*)(p + 8);
    v16bf r;
#pragma unroll
    for (int i = 0; i < 8; ++i) { r[i] = lo[i]; r[8 + i] = hi[i]; }
    return r;
}

// LDS byte offset of a __shared__ object (for async-to-LDS VDST operand)
typedef __attribute__((address_space(3))) char lds_char;
__device__ inline unsigned lds_addr_of(const void* p) {
    return (unsigned)(uintptr_t)(lds_char*)p;
}

// ---------------------------------------------------------------------------
// f32 -> bf16 elementwise convert (grid-stride)
// ---------------------------------------------------------------------------
__global__ void f32_to_bf16_kernel(const float* __restrict__ in,
                                   bf16* __restrict__ out, size_t n) {
    size_t i = (size_t)blockIdx.x * blockDim.x + threadIdx.x;
    size_t stride = (size_t)gridDim.x * blockDim.x;
    for (; i < n; i += stride) out[i] = (bf16)in[i];
}

// ---------------------------------------------------------------------------
// C[M,N] = A[M,K] @ W[N,K]^T   (bf16 in, f32 accum, bf16 or f32 out)
// One wave computes FOUR 16x16 C tiles (16 rows x 64 cols). Ping-pong double
// buffering, no copies / no in-loop guards (requires K % 64 == 0): each
// buffer's refill hides under the other buffer's 4 WMMAs.
// ---------------------------------------------------------------------------
struct Frags {
    v16bf a;
    v16bf b[4];
};

__device__ inline void load_frags(const bf16* arow, const bf16* const* wrow,
                                  int kk, int g, Frags& f) {
    f.a = load_a_frag(arow + kk + 16 * g);
#pragma unroll
    for (int s = 0; s < 4; ++s)
        f.b[s] = *(const v16bf*)(wrow[s] + kk + 16 * g);
}

__global__ __launch_bounds__(256) void gemm_bf16_kernel(
    const bf16* __restrict__ A, const bf16* __restrict__ W,
    int N, int K, bf16* out_bf, float* out_f) {
    const int lane   = threadIdx.x & 31;
    const int wave   = threadIdx.x >> 5;
    const int ntile0 = (blockIdx.x * 8 + wave) * 4;
    const int mtile  = blockIdx.y;
    if (ntile0 * 16 >= N) return;

    const int g   = lane >> 4;
    const int n16 = lane & 15;
    const bf16* arow = A + (size_t)(mtile * 16 + n16) * K;
    const bf16* wrow[4];
#pragma unroll
    for (int s = 0; s < 4; ++s)
        wrow[s] = W + (size_t)((ntile0 + s) * 16 + n16) * K;

    v8f c[4];
#pragma unroll
    for (int s = 0; s < 4; ++s) { v8f z = {}; c[s] = z; }

    Frags f0, f1;
    load_frags(arow, wrow, 0, g, f0);
    load_frags(arow, wrow, 32, g, f1);
    int kk = 0;
    for (; kk + 64 < K; kk += 64) {
        __builtin_prefetch(arow + kk + 1024, 0, 3);
#pragma unroll
        for (int s = 0; s < 4; ++s)
            c[s] = __builtin_amdgcn_wmma_f32_16x16x32_bf16(false, f0.a, false,
                                                           f0.b[s], (short)0,
                                                           c[s], false, false);
        load_frags(arow, wrow, kk + 64, g, f0);
#pragma unroll
        for (int s = 0; s < 4; ++s)
            c[s] = __builtin_amdgcn_wmma_f32_16x16x32_bf16(false, f1.a, false,
                                                           f1.b[s], (short)0,
                                                           c[s], false, false);
        load_frags(arow, wrow, kk + 96, g, f1);
    }
    // epilogue: consume the last two buffers
#pragma unroll
    for (int s = 0; s < 4; ++s)
        c[s] = __builtin_amdgcn_wmma_f32_16x16x32_bf16(false, f0.a, false,
                                                       f0.b[s], (short)0,
                                                       c[s], false, false);
#pragma unroll
    for (int s = 0; s < 4; ++s)
        c[s] = __builtin_amdgcn_wmma_f32_16x16x32_bf16(false, f1.a, false,
                                                       f1.b[s], (short)0,
                                                       c[s], false, false);

    // C layout: vgpr r, lanes0-15: M=r; lanes16-31: M=r+8; N = lane%16
#pragma unroll
    for (int s = 0; s < 4; ++s) {
        int col = (ntile0 + s) * 16 + n16;
#pragma unroll
        for (int r = 0; r < 8; ++r) {
            int row = mtile * 16 + r + 8 * g;
            if (out_f) out_f[(size_t)row * N + col] = c[s][r];
            else       out_bf[(size_t)row * N + col] = (bf16)c[s][r];
        }
    }
}

// ---------------------------------------------------------------------------
// In-place RoPE on bf16 [tokens, n_heads, 128]; one thread per rotation pair.
// ---------------------------------------------------------------------------
__global__ void rope_kernel(bf16* __restrict__ X, const int* __restrict__ pos_ids,
                            int n_heads) {
    const long long total = (long long)B_ * S_ * n_heads * 64;
    long long t = (long long)blockIdx.x * blockDim.x + threadIdx.x;
    if (t >= total) return;
    int j    = (int)(t & 63);
    int head = (int)((t >> 6) % n_heads);
    int tok  = (int)(t / (64LL * n_heads));

    bf16* q = X + ((size_t)tok * n_heads + head) * HD;
    float x1 = (float)q[j];
    float x2 = (float)q[j + 64];
    float pos = (float)pos_ids[tok];
    float inv_freq = __powf(10000.0f, -(float)(2 * j) / 128.0f);
    float ang = pos * inv_freq;
    float c = __cosf(ang), s = __sinf(ang);
    q[j]      = (bf16)(x1 * c - x2 * s);
    q[j + 64] = (bf16)(x2 * c + x1 * s);
}

// ---------------------------------------------------------------------------
// Flash attention, sliding window, GQA. One wave per (b, head, 16-query tile).
// 4 waves / block. V chunk staged into wave-private LDS with
// GLOBAL_LOAD_ASYNC_TO_LDS_B128 (async copy, ASYNCcnt-fenced).
// ---------------------------------------------------------------------------
__global__ __launch_bounds__(128) void attn_kernel(
    const bf16* __restrict__ Q, const bf16* __restrict__ K,
    const bf16* __restrict__ V, bf16* __restrict__ O) {
    __shared__ __align__(128) bf16 smem[4 * (32 * HD + 16 * 32)];
    const int lane = threadIdx.x & 31;
    const int wave = threadIdx.x >> 5;
    bf16* Vlin = smem + wave * (32 * HD + 16 * 32);  // [32 keys][128 dims]
    bf16* Pm   = Vlin + 32 * HD;                     // [16 rows][32 keys]
    const unsigned vlin_lds = lds_addr_of(Vlin);

    const int hg    = blockIdx.x % (NH / 4);
    const int qt    = (blockIdx.x / (NH / 4)) % (S_ / 16);
    const int b     = blockIdx.x / ((NH / 4) * (S_ / 16));
    const int head  = hg * 4 + wave;
    const int kvh   = head / (NH / NKV);
    const int qbase = qt * 16;

    const int g   = lane >> 4;
    const int n16 = lane & 15;

    // Q fragments for all four K-dim steps (head dim 128 = 4 x 32)
    const bf16* qrow = Q + ((size_t)(b * S_ + qbase + n16) * NH + head) * HD;
    v16bf aq[4];
#pragma unroll
    for (int t = 0; t < 4; ++t) aq[t] = load_a_frag(qrow + 32 * t + 16 * g);

    v8f acc[8];
#pragma unroll
    for (int i = 0; i < 8; ++i) { v8f z = {}; acc[i] = z; }
    float mrow[8], lrow[8];
#pragma unroll
    for (int r = 0; r < 8; ++r) { mrow[r] = -1e30f; lrow[r] = 0.0f; }

    const float scale = 0.08838834764831845f;  // 1/sqrt(128)

    int kstart = qbase - (WIN - 1);
    if (kstart < 0) kstart = 0;
    kstart &= ~31;

    for (int kb = kstart; kb <= qbase + 15; kb += 32) {
        // ---- stage V chunk into LDS via async copy: Vlin[key][dim] ----
#pragma unroll
        for (int i = 0; i < 16; ++i) {
            int c = i * 32 + lane;      // 512 x 16B chunks
            int key = c >> 4;           // 0..31
            int dchunk = c & 15;        // 0..15
            int kpos = kb + key;
            if (kpos < S_) {
                const bf16* gsrc = V + ((size_t)(b * S_ + kpos) * NKV + kvh) * HD
                                     + dchunk * 8;
                unsigned ldsaddr = vlin_lds + (unsigned)(key * 256 + dchunk * 16);
                asm volatile("global_load_async_to_lds_b128 %0, %1, off"
                             :: "v"(ldsaddr),
                                "v"((unsigned long long)(uintptr_t)gsrc)
                             : "memory");
            } else {
                v8bf z = {};
                *(v8bf*)(Vlin + key * HD + dchunk * 8) = z;  // zero-fill OOB keys
            }
        }

        // ---- S = Q @ K^T for the two 16-key tiles of this chunk ----
        float sc[2][8];
#pragma unroll
        for (int t2 = 0; t2 < 2; ++t2) {
            int kcb  = kb + t2 * 16;
            int kpos = kcb + n16;  // B-matrix column n = key
            const bf16* krow = K + ((size_t)(b * S_ + kpos) * NKV + kvh) * HD;
            v16bf bk[4];
#pragma unroll
            for (int t = 0; t < 4; ++t) {            // all 4 fragments in flight
                v16bf z = {};
                bk[t] = (kpos < S_) ? *(const v16bf*)(krow + 32 * t + 16 * g) : z;
            }
            v8f s = {};
#pragma unroll
            for (int t = 0; t < 4; ++t)
                s = __builtin_amdgcn_wmma_f32_16x16x32_bf16(false, aq[t], false,
                                                            bk[t], (short)0, s,
                                                            false, false);
#pragma unroll
            for (int r = 0; r < 8; ++r) {
                int row  = r + 8 * g;
                int qpos = qbase + row;
                int kp   = kcb + n16;
                bool ok  = (kp <= qpos) && ((qpos - kp) < WIN) && (kp < S_);
                sc[t2][r] = ok ? s[r] * scale : -1e30f;
            }
        }

        // ---- online softmax (row stats via 16-lane butterfly) ----
#pragma unroll
        for (int r = 0; r < 8; ++r) {
            float cm = fmaxf(sc[0][r], sc[1][r]);
            for (int m = 1; m <= 8; m <<= 1) cm = fmaxf(cm, __shfl_xor(cm, m, 32));
            float m2    = fmaxf(mrow[r], cm);
            float alpha = (mrow[r]  > -1e29f) ? __expf(mrow[r]  - m2) : 0.0f;
            float p0    = (sc[0][r] > -1e29f) ? __expf(sc[0][r] - m2) : 0.0f;
            float p1    = (sc[1][r] > -1e29f) ? __expf(sc[1][r] - m2) : 0.0f;
            float rs = p0 + p1;
            for (int m = 1; m <= 8; m <<= 1) rs += __shfl_xor(rs, m, 32);
            lrow[r] = lrow[r] * alpha + rs;
            mrow[r] = m2;
#pragma unroll
            for (int dt = 0; dt < 8; ++dt) acc[dt][r] *= alpha;
            int row = r + 8 * g;
            Pm[row * 32 + n16]      = (bf16)p0;
            Pm[row * 32 + 16 + n16] = (bf16)p1;
        }

        // make async-staged V and P stores visible to this wave
        asm volatile("s_wait_asynccnt 0x0" ::: "memory");
        asm volatile("s_wait_dscnt 0x0" ::: "memory");

        // ---- acc += P @ V  (K = 32 keys, 8 tiles across head dim) ----
        v16bf ap = load_a_frag(Pm + n16 * 32 + 16 * g);
#pragma unroll
        for (int dt = 0; dt < 8; ++dt) {
            int d = dt * 16 + n16;        // B column = head dim
            v16bf bv;
#pragma unroll
            for (int e = 0; e < 16; ++e)  // K = 16 keys at stride HD (strided ds)
                bv[e] = Vlin[(size_t)(16 * g + e) * HD + d];
            acc[dt] = __builtin_amdgcn_wmma_f32_16x16x32_bf16(false, ap, false, bv,
                                                              (short)0, acc[dt],
                                                              false, false);
        }
    }

    // ---- normalize and store bf16 output [token, head, dim] ----
#pragma unroll
    for (int r = 0; r < 8; ++r) {
        float inv = (lrow[r] > 0.0f) ? 1.0f / lrow[r] : 0.0f;
        int qpos = qbase + r + 8 * g;
        bf16* orow = O + ((size_t)(b * S_ + qpos) * NH + head) * HD;
#pragma unroll
        for (int dt = 0; dt < 8; ++dt)
            orow[dt * 16 + n16] = (bf16)(acc[dt][r] * inv);
    }
}

// ---------------------------------------------------------------------------
extern "C" void kernel_launch(void* const* d_in, const int* in_sizes, int n_in,
                              void* d_out, int out_size, void* d_ws, size_t ws_size,
                              hipStream_t stream) {
    (void)in_sizes; (void)n_in; (void)out_size; (void)ws_size;
    const float* hs = (const float*)d_in[0];
    const float* Wq = (const float*)d_in[1];
    const float* Wk = (const float*)d_in[2];
    const float* Wv = (const float*)d_in[3];
    const float* Wo = (const float*)d_in[4];
    const int*  pos = (const int*)d_in[5];

    char* cur = (char*)d_ws;
    auto alloc = [&](size_t elems) -> bf16* {
        bf16* r = (bf16*)cur;
        cur += ((elems * sizeof(bf16)) + 255) & ~(size_t)255;
        return r;
    };
    const size_t T = (size_t)B_ * S_;  // 4096 tokens
    bf16* Xbf = alloc(T * H_);
    bf16* Wqb = alloc((size_t)NH * HD * H_);
    bf16* Wkb = alloc((size_t)NKV * HD * H_);
    bf16* Wvb = alloc((size_t)NKV * HD * H_);
    bf16* Wob = alloc((size_t)H_ * NH * HD);
    bf16* Qb  = alloc(T * NH * HD);
    bf16* Kb  = alloc(T * NKV * HD);
    bf16* Vb  = alloc(T * NKV * HD);
    bf16* Ab  = alloc(T * NH * HD);

    // 1) convert everything to bf16
    f32_to_bf16_kernel<<<2048, 256, 0, stream>>>(hs, Xbf, T * H_);
    f32_to_bf16_kernel<<<2048, 256, 0, stream>>>(Wq, Wqb, (size_t)NH * HD * H_);
    f32_to_bf16_kernel<<<1024, 256, 0, stream>>>(Wk, Wkb, (size_t)NKV * HD * H_);
    f32_to_bf16_kernel<<<1024, 256, 0, stream>>>(Wv, Wvb, (size_t)NKV * HD * H_);
    f32_to_bf16_kernel<<<2048, 256, 0, stream>>>(Wo, Wob, (size_t)H_ * NH * HD);

    // 2) QKV projections (WMMA GEMMs), bf16 epilogue   (K = 4096, % 64 == 0)
    gemm_bf16_kernel<<<dim3((NH * HD) / 512, T / 16), 256, 0, stream>>>(
        Xbf, Wqb, NH * HD, H_, Qb, nullptr);
    gemm_bf16_kernel<<<dim3((NKV * HD) / 512, T / 16), 256, 0, stream>>>(
        Xbf, Wkb, NKV * HD, H_, Kb, nullptr);
    gemm_bf16_kernel<<<dim3((NKV * HD) / 512, T / 16), 256, 0, stream>>>(
        Xbf, Wvb, NKV * HD, H_, Vb, nullptr);

    // 3) RoPE in place on Q and K
    {
        long long nq = (long long)T * NH * 64;
        long long nk = (long long)T * NKV * 64;
        rope_kernel<<<(unsigned)((nq + 255) / 256), 256, 0, stream>>>(Qb, pos, NH);
        rope_kernel<<<(unsigned)((nk + 255) / 256), 256, 0, stream>>>(Kb, pos, NKV);
    }

    // 4) sliding-window flash attention -> bf16 [token, head, dim]
    attn_kernel<<<B_ * (S_ / 16) * (NH / 4), 128, 0, stream>>>(Qb, Kb, Vb, Ab);

    // 5) output projection, f32 epilogue straight to d_out   (K = 4096)
    gemm_bf16_kernel<<<dim3(H_ / 512, T / 16), 256, 0, stream>>>(
        Ab, Wob, H_, NH * HD, nullptr, (float*)d_out);
}
